// MultiheadAttention_9474697855207
// MI455X (gfx1250) — matrix operand order repaired and checked by reading, not compiled, not run on hardware
//
#include <hip/hip_runtime.h>
#include <hip/hip_bf16.h>

// ---------------------------------------------------------------------------
// MultiheadAttention for MI455X (gfx1250): B=2, S=2048, D=1024, H=16, DH=64
// Pre-convert all operands to f16 once; GEMMs via v_wmma_f32_16x16x32_f16
// (wave32, 64x64 register blocking, pure b128-load inner loops).
// Attention: flash-style, async LDS staging (global_load_async_to_lds_b128).
// ---------------------------------------------------------------------------

typedef __attribute__((ext_vector_type(16))) _Float16 v16h;
typedef __attribute__((ext_vector_type(8)))  _Float16 v8h;
typedef __attribute__((ext_vector_type(8)))  float    v8f;
typedef __attribute__((ext_vector_type(4)))  float    v4f;

#define Bdim 2
#define Sdim 2048
#define Ddim 1024
#define Hdim 16
#define DHdim 64
#define Mrows (Bdim * Sdim)   // 4096

static __device__ inline v8f wmma_f16(v16h a, v16h b, v8f c) {
    return __builtin_amdgcn_wmma_f32_16x16x32_f16(
        /*neg_a=*/false, a, /*neg_b=*/false, b,
        /*c_mod=*/(short)0, c, /*reuse_a=*/false, /*reuse_b=*/false);
}

// --- CDNA5 async copy: 16 bytes global -> LDS, tracked by ASYNCcnt ---------
static __device__ inline void async_copy_b128(void* lds, const void* gaddr) {
    // Low 32 bits of a flat LDS address are the LDS byte offset (ISA 10.2).
    unsigned l = (unsigned)(unsigned long long)lds;
    asm volatile("global_load_async_to_lds_b128 %0, %1, off"
                 :: "v"(l), "v"(gaddr) : "memory");
}
static __device__ inline void wait_asynccnt0() {
    asm volatile("s_wait_asynccnt 0x0" ::: "memory");
}

// A-fragment (16x32 f16, MxK): lane holds row m = lane%16.
// Per lane the 16 halves are one contiguous run:
//   elements 0..15 = K [k0 + 16*hf .. +15]
static __device__ inline v16h load_afrag_f16(const _Float16* row, int k0, int hf) {
    v8h lo = *(const v8h*)(row + k0 + 16 * hf);
    v8h hi = *(const v8h*)(row + k0 + 16 * hf + 8);
    v16h a;
#pragma unroll
    for (int i = 0; i < 8; ++i) { a[i] = lo[i]; a[8 + i] = hi[i]; }
    return a;
}

// B-fragment (32x16 f16, KxN): lane holds column n = lane%16,
// K range = 16*(lane/16) .. +15, contiguous in memory per lane.
static __device__ inline v16h load_bfrag_f16(const _Float16* p) {
    const v8h* q = (const v8h*)p;
    v8h lo = q[0], hi = q[1];
    v16h b;
#pragma unroll
    for (int i = 0; i < 8; ++i) { b[i] = lo[i]; b[8 + i] = hi[i]; }
    return b;
}

// ---------------------------------------------------------------------------
// Kernel 0: bulk f32 -> f16 conversion (8 elements / thread, b128 in/out)
// ---------------------------------------------------------------------------
__global__ __launch_bounds__(256) void cvt_f32_to_f16_kernel(
    const float* __restrict__ src, _Float16* __restrict__ dst, int n) {
    int i = (blockIdx.x * 256 + threadIdx.x) * 8;
    if (i >= n) return;
    v4f a = *(const v4f*)(src + i);
    v4f b = *(const v4f*)(src + i + 4);
    v8h o;
#pragma unroll
    for (int j = 0; j < 4; ++j) { o[j] = (_Float16)a[j]; o[4 + j] = (_Float16)b[j]; }
    *(v8h*)(dst + i) = o;
}

// ---------------------------------------------------------------------------
// Kernel 1: fused QKV projection.  C[4096,1024] = X @ W^T + bias (all f16 in)
// Each wave computes a 64x64 tile: 4 A-frags x 4 B-frags -> 16 WMMAs/k-step.
// Writes f16 in head layout [B,H,S,DH]; Q pre-scaled by 1/sqrt(DH).
// grid = (16, 16, 3), block = 128 (4 waves)
// ---------------------------------------------------------------------------
__global__ __launch_bounds__(128) void qkv_proj_kernel(
    const _Float16* __restrict__ xq16, const _Float16* __restrict__ xkv16,
    const _Float16* __restrict__ Wq16, const float* __restrict__ bq,
    const _Float16* __restrict__ Wk16, const float* __restrict__ bk,
    const _Float16* __restrict__ Wv16, const float* __restrict__ bv,
    _Float16* __restrict__ qh, _Float16* __restrict__ kh,
    _Float16* __restrict__ vh) {
    const int which = blockIdx.z;  // 0=Q, 1=K, 2=V
    const _Float16* X    = (which == 0) ? xq16 : xkv16;
    const _Float16* W    = (which == 0) ? Wq16 : ((which == 1) ? Wk16 : Wv16);
    const float*    bias = (which == 0) ? bq : ((which == 1) ? bk : bv);
    _Float16*       Out  = (which == 0) ? qh : ((which == 1) ? kh : vh);

    const int wave = threadIdx.x >> 5;
    const int lane = threadIdx.x & 31;
    const int hf   = lane >> 4;
    const int ln   = lane & 15;

    const int m0 = (blockIdx.x * 4 + wave) * 64;  // 64-row wave tile
    const int n0 = blockIdx.y * 64;               // 64-col wave tile

    const _Float16* arow[4];
    const _Float16* brow[4];
#pragma unroll
    for (int i = 0; i < 4; ++i) {
        arow[i] = X + (size_t)(m0 + i * 16 + ln) * Ddim;
        brow[i] = W + (size_t)(n0 + i * 16 + ln) * Ddim;
    }

    v8f acc[4][4];
#pragma unroll
    for (int i = 0; i < 4; ++i)
#pragma unroll
        for (int j = 0; j < 4; ++j) acc[i][j] = (v8f){};

#pragma unroll 2
    for (int k0 = 0; k0 < Ddim; k0 += 32) {
        v16h a[4], b[4];
#pragma unroll
        for (int i = 0; i < 4; ++i) a[i] = load_afrag_f16(arow[i], k0, hf);
#pragma unroll
        for (int j = 0; j < 4; ++j) b[j] = load_bfrag_f16(brow[j] + k0 + 16 * hf);
#pragma unroll
        for (int i = 0; i < 4; ++i)
#pragma unroll
            for (int j = 0; j < 4; ++j) acc[i][j] = wmma_f16(a[i], b[j], acc[i][j]);
    }

    const float qscale = (which == 0) ? 0.125f : 1.0f;  // 1/sqrt(64)
#pragma unroll
    for (int j = 0; j < 4; ++j) {
        const int ncol = n0 + j * 16 + ln;
        const float bcol = bias[ncol];
        const int h  = ncol >> 6;
        const int dh = ncol & 63;
#pragma unroll
        for (int i = 0; i < 4; ++i) {
#pragma unroll
            for (int r = 0; r < 8; ++r) {
                int m  = m0 + i * 16 + r + 8 * hf;  // global row in [0,4096)
                int bb = m >> 11;
                int s  = m & (Sdim - 1);
                float val = (acc[i][j][r] + bcol) * qscale;
                Out[(((size_t)(bb * Hdim + h)) * Sdim + s) * DHdim + dh] =
                    (_Float16)val;
            }
        }
    }
}

// ---------------------------------------------------------------------------
// Kernel 2: flash attention.  4 waves share one (b,h); each owns 16 queries.
// Per 32-key iteration: K tile async-copied to LDS, V tile staged transposed
// in LDS (double-buffered), 8 WMMAs per wave.
// grid = (B*H=32, S/64=32), block = 128
// ---------------------------------------------------------------------------
__global__ __launch_bounds__(128) void attn_kernel(
    const _Float16* __restrict__ qh, const _Float16* __restrict__ kh,
    const _Float16* __restrict__ vh, _Float16* __restrict__ aout) {
    __shared__ _Float16 Kt[2][32 * DHdim];   // [key][d]  4KB x2 (async-filled)
    __shared__ _Float16 Vt[2][DHdim * 32];   // [d][key]  4KB x2 (transposed)
    __shared__ _Float16 Pst[4][16 * 32];     // per-wave P staging

    const int tid  = threadIdx.x;
    const int wave = tid >> 5;
    const int lane = tid & 31;
    const int hf   = lane >> 4;
    const int ln   = lane & 15;

    const int bh = blockIdx.x;  // 0..31
    const int b  = bh >> 4;
    const int h  = bh & 15;

    const _Float16* Q = qh + (size_t)bh * Sdim * DHdim;
    const _Float16* K = kh + (size_t)bh * Sdim * DHdim;
    const _Float16* V = vh + (size_t)bh * Sdim * DHdim;

    const int q0 = blockIdx.y * 64 + wave * 16;
    const _Float16* qrow = Q + (size_t)(q0 + ln) * DHdim;
    const v16h qa0 = load_afrag_f16(qrow, 0,  hf);   // d = 0..31
    const v16h qa1 = load_afrag_f16(qrow, 32, hf);   // d = 32..63

    v8f o0 = {}, o1 = {}, o2 = {}, o3 = {};
    float mrow[8], lrow[8];
#pragma unroll
    for (int r = 0; r < 8; ++r) { mrow[r] = -1e30f; lrow[r] = 0.0f; }

    _Float16* P = Pst[wave];

    // V staging role for this thread: key row + 16-wide d slice
    const int vk = tid & 31;          // key row 0..31
    const int vd = (tid >> 5) * 16;   // d offset 0,16,32,48

    // ---- stage tile j into buffer buf ----
    auto stage = [&](int j0, int buf) {
        // K tile: raw async copy, 4KB = 256 x 16B chunks, 128 threads x 2
        const _Float16* kt = K + (size_t)j0 * DHdim;
        async_copy_b128(&Kt[buf][(size_t)tid * 8],         kt + (size_t)tid * 8);
        async_copy_b128(&Kt[buf][(size_t)(tid + 128) * 8], kt + (size_t)(tid + 128) * 8);
        // V tile: transpose while staging: Vt[d][k] = V[j0+k][d]
        const v8h* src = (const v8h*)(V + (size_t)(j0 + vk) * DHdim + vd);
        v8h x0 = src[0], x1 = src[1];
#pragma unroll
        for (int i = 0; i < 8; ++i) {
            Vt[buf][(vd + i) * 32 + vk]     = x0[i];
            Vt[buf][(vd + 8 + i) * 32 + vk] = x1[i];
        }
    };

    stage(0, 0);
    wait_asynccnt0();
    __syncthreads();

    for (int j0 = 0; j0 < Sdim; j0 += 32) {
        const int buf = (j0 >> 5) & 1;

        // ---- logits: two 16x16 tiles over d=64 (4 WMMAs), K from LDS ----
        v8f s0 = {}, s1 = {};
        {
            const _Float16* k0row = &Kt[buf][(ln) * DHdim];
            const _Float16* k1row = &Kt[buf][(16 + ln) * DHdim];
            s0 = wmma_f16(qa0, load_bfrag_f16(k0row + 16 * hf), s0);
            s0 = wmma_f16(qa1, load_bfrag_f16(k0row + 32 + 16 * hf), s0);
            s1 = wmma_f16(qa0, load_bfrag_f16(k1row + 16 * hf), s1);
            s1 = wmma_f16(qa1, load_bfrag_f16(k1row + 32 + 16 * hf), s1);
        }

        // ---- online softmax (row r lives in VGPR r across a 16-lane half)
#pragma unroll
        for (int r = 0; r < 8; ++r) {
            float t = fmaxf(s0[r], s1[r]);
            t = fmaxf(t, __shfl_xor(t, 1, 32));
            t = fmaxf(t, __shfl_xor(t, 2, 32));
            t = fmaxf(t, __shfl_xor(t, 4, 32));
            t = fmaxf(t, __shfl_xor(t, 8, 32));
            float mnew  = fmaxf(mrow[r], t);
            float alpha = __expf(mrow[r] - mnew);
            mrow[r] = mnew;
            float p0 = __expf(s0[r] - mnew);
            float p1 = __expf(s1[r] - mnew);
            s0[r] = p0; s1[r] = p1;
            float ps = p0 + p1;
            ps += __shfl_xor(ps, 1, 32);
            ps += __shfl_xor(ps, 2, 32);
            ps += __shfl_xor(ps, 4, 32);
            ps += __shfl_xor(ps, 8, 32);
            lrow[r] = lrow[r] * alpha + ps;
            o0[r] *= alpha; o1[r] *= alpha; o2[r] *= alpha; o3[r] *= alpha;
        }

        // ---- stage P (C-layout -> row-major LDS), reload as A-fragment ----
#pragma unroll
        for (int r = 0; r < 8; ++r) {
            int rr = r + 8 * hf;
            P[rr * 32 + ln]      = (_Float16)s0[r];
            P[rr * 32 + 16 + ln] = (_Float16)s1[r];
        }
        v16h pa = load_afrag_f16(P + ln * 32, 0, hf);

        // ---- P @ V: 4 output tiles, B-frags contiguous from transposed LDS
        o0 = wmma_f16(pa, load_bfrag_f16(&Vt[buf][(0  + ln) * 32 + 16 * hf]), o0);
        o1 = wmma_f16(pa, load_bfrag_f16(&Vt[buf][(16 + ln) * 32 + 16 * hf]), o1);
        o2 = wmma_f16(pa, load_bfrag_f16(&Vt[buf][(32 + ln) * 32 + 16 * hf]), o2);
        o3 = wmma_f16(pa, load_bfrag_f16(&Vt[buf][(48 + ln) * 32 + 16 * hf]), o3);

        // ---- prefetch next tile into the other buffer ----
        if (j0 + 32 < Sdim) stage(j0 + 32, buf ^ 1);
        wait_asynccnt0();
        __syncthreads();
    }

    // ---- normalize and write merged-head layout [B,S,D] f16 ----
#pragma unroll
    for (int r = 0; r < 8; ++r) {
        float inv = 1.0f / lrow[r];
        int m = q0 + r + 8 * hf;
        size_t base = ((size_t)b * Sdim + m) * Ddim + h * DHdim + ln;
        aout[base + 0]  = (_Float16)(o0[r] * inv);
        aout[base + 16] = (_Float16)(o1[r] * inv);
        aout[base + 32] = (_Float16)(o2[r] * inv);
        aout[base + 48] = (_Float16)(o3[r] * inv);
    }
}

// ---------------------------------------------------------------------------
// Kernel 3: output projection.  out[4096,1024] f32 = A(f16) @ W_out^T + b_out
// 64x64 per wave, grid = (16, 16), block = 128
// ---------------------------------------------------------------------------
__global__ __launch_bounds__(128) void out_proj_kernel(
    const _Float16* __restrict__ A, const _Float16* __restrict__ W16,
    const float* __restrict__ bias, float* __restrict__ out) {
    const int wave = threadIdx.x >> 5;
    const int lane = threadIdx.x & 31;
    const int hf   = lane >> 4;
    const int ln   = lane & 15;

    const int m0 = (blockIdx.x * 4 + wave) * 64;
    const int n0 = blockIdx.y * 64;

    const _Float16* arow[4];
    const _Float16* brow[4];
#pragma unroll
    for (int i = 0; i < 4; ++i) {
        arow[i] = A + (size_t)(m0 + i * 16 + ln) * Ddim;
        brow[i] = W16 + (size_t)(n0 + i * 16 + ln) * Ddim;
    }

    v8f acc[4][4];
#pragma unroll
    for (int i = 0; i < 4; ++i)
#pragma unroll
        for (int j = 0; j < 4; ++j) acc[i][j] = (v8f){};

#pragma unroll 2
    for (int k0 = 0; k0 < Ddim; k0 += 32) {
        v16h a[4], b[4];
#pragma unroll
        for (int i = 0; i < 4; ++i) a[i] = load_afrag_f16(arow[i], k0, hf);
#pragma unroll
        for (int j = 0; j < 4; ++j) b[j] = load_bfrag_f16(brow[j] + k0 + 16 * hf);
#pragma unroll
        for (int i = 0; i < 4; ++i)
#pragma unroll
            for (int j = 0; j < 4; ++j) acc[i][j] = wmma_f16(a[i], b[j], acc[i][j]);
    }

#pragma unroll
    for (int j = 0; j < 4; ++j) {
        const int ncol = n0 + j * 16 + ln;
        const float bcol = bias[ncol];
#pragma unroll
        for (int i = 0; i < 4; ++i)
#pragma unroll
            for (int r = 0; r < 8; ++r) {
                int m = m0 + i * 16 + r + 8 * hf;
                out[(size_t)m * Ddim + ncol] = acc[i][j][r] + bcol;
            }
    }
}

// ---------------------------------------------------------------------------
extern "C" void kernel_launch(void* const* d_in, const int* in_sizes, int n_in,
                              void* d_out, int out_size, void* d_ws,
                              size_t ws_size, hipStream_t stream) {
    const float* xq   = (const float*)d_in[0];
    const float* xkv  = (const float*)d_in[1];
    const float* Wq   = (const float*)d_in[2];
    const float* bq   = (const float*)d_in[3];
    const float* Wk   = (const float*)d_in[4];
    const float* bk   = (const float*)d_in[5];
    const float* Wv   = (const float*)d_in[6];
    const float* bv   = (const float*)d_in[7];
    const float* Wout = (const float*)d_in[8];
    const float* bout = (const float*)d_in[9];

    const size_t nx = (size_t)Mrows * Ddim;  // 4M elements (x / per-head bufs)
    const size_t nw = (size_t)Ddim * Ddim;   // 1M elements (weights)

    _Float16* p = (_Float16*)d_ws;
    _Float16* xq16  = p; p += nx;
    _Float16* xkv16 = p; p += nx;
    _Float16* wq16  = p; p += nw;
    _Float16* wk16  = p; p += nw;
    _Float16* wv16  = p; p += nw;
    _Float16* wo16  = p; p += nw;
    _Float16* qh    = p; p += nx;
    _Float16* kh    = p; p += nx;
    _Float16* vh    = p; p += nx;
    _Float16* aout  = p; p += nx;

    // ---- stage 0: one-time f32 -> f16 conversions ----
    cvt_f32_to_f16_kernel<<<nx / 2048, 256, 0, stream>>>(xq,   xq16,  (int)nx);
    cvt_f32_to_f16_kernel<<<nx / 2048, 256, 0, stream>>>(xkv,  xkv16, (int)nx);
    cvt_f32_to_f16_kernel<<<nw / 2048, 256, 0, stream>>>(Wq,   wq16,  (int)nw);
    cvt_f32_to_f16_kernel<<<nw / 2048, 256, 0, stream>>>(Wk,   wk16,  (int)nw);
    cvt_f32_to_f16_kernel<<<nw / 2048, 256, 0, stream>>>(Wv,   wv16,  (int)nw);
    cvt_f32_to_f16_kernel<<<nw / 2048, 256, 0, stream>>>(Wout, wo16,  (int)nw);

    // ---- stage 1: QKV projections ----
    qkv_proj_kernel<<<dim3(Mrows / 256, Ddim / 64, 3), 128, 0, stream>>>(
        xq16, xkv16, wq16, bq, wk16, bk, wv16, bv, qh, kh, vh);

    // ---- stage 2: flash attention ----
    attn_kernel<<<dim3(Bdim * Hdim, Sdim / 64), 128, 0, stream>>>(
        qh, kh, vh, aout);

    // ---- stage 3: output projection ----
    out_proj_kernel<<<dim3(Mrows / 256, Ddim / 64), 128, 0, stream>>>(
        aout, wo16, bout, (float*)d_out);
}
